// MultiHeadAttention_36155034698425
// MI455X (gfx1250) — matrix operand order repaired and checked by reading, not compiled, not run on hardware
//
#include <hip/hip_runtime.h>
#include <hip/hip_bf16.h>

// ---------------- problem constants ----------------
constexpr int Bb = 2;
constexpr int Tt = 2048;
constexpr int Dd = 1024;
constexpr int Hh = 16;
constexpr int HS = 64;                        // head size
constexpr int BT = Bb * Tt;                   // 4096 rows in (b,t) space
constexpr size_t BTD = (size_t)Bb * Tt * Dd;  // 4,194,304 elements
constexpr size_t DD  = (size_t)Dd * Dd;       // 1,048,576

typedef __attribute__((ext_vector_type(16))) _Float16 v16h;
typedef __attribute__((ext_vector_type(8)))  _Float16 v8h;
typedef __attribute__((ext_vector_type(8)))  float    v8f;

// ------------- gfx1250 async copy-to-LDS (guarded) -------------
#if defined(__has_builtin)
#  if __has_builtin(__builtin_amdgcn_global_load_async_to_lds_b128)
#    define HAVE_ASYNC_LDS 1
#  endif
#endif
#ifndef HAVE_ASYNC_LDS
#  define HAVE_ASYNC_LDS 0
#endif

#if HAVE_ASYNC_LDS
// builtin signature (from hipcc diagnostic): param0 = v4i addrspace(1)*,
// then LDS v4i addrspace(3)*, imm offset, imm cpol
typedef int v4i_t __attribute__((vector_size(16)));
typedef __attribute__((address_space(1))) v4i_t gas_v4i;
typedef __attribute__((address_space(3))) v4i_t las_v4i;
#endif

static __device__ __forceinline__ void wait_async0() {
#if HAVE_ASYNC_LDS
#  if __has_builtin(__builtin_amdgcn_s_wait_asynccnt)
  __builtin_amdgcn_s_wait_asynccnt(0);
#  else
  asm volatile("s_wait_asynccnt 0x0" ::: "memory");
#  endif
#endif
}

static __device__ __forceinline__ v8f wmma16(v16h a, v16h b, v8f c) {
  // D = A(16x32) * B(32x16) + C(16x16), f32 accumulate
  return __builtin_amdgcn_wmma_f32_16x16x32_f16(false, a, false, b, (short)0, c, false, false);
}

// ---- A fragment (16x32, row-major f16, lane = row) ----
// element j -> k = j + 16*half, half = lane>>4
static __device__ __forceinline__ v16h load_a_f16(const _Float16* base, int ld) {
  const int lane = threadIdx.x & 31;
  const int m = lane & 15, hf = lane >> 4;
  const _Float16* p0 = base + (size_t)m * ld + 16 * hf;
  const _Float16* p1 = p0 + 8;
  v16h a;
#pragma unroll
  for (int j = 0; j < 8; ++j) a[j] = p0[j];
#pragma unroll
  for (int j = 0; j < 8; ++j) a[8 + j] = p1[j];
  return a;
}

// ---- B fragment from W stored [N x K] row-major f16 (computes A * W^T) ----
// lane = column n (= W row), element j -> k = j + 16*half : contiguous load
static __device__ __forceinline__ v16h load_bt_f16(const _Float16* base, int ld) {
  const int lane = threadIdx.x & 31;
  const int n = lane & 15, hf = lane >> 4;
  const _Float16* p = base + (size_t)n * ld + 16 * hf;
  v16h b;
#pragma unroll
  for (int j = 0; j < 16; ++j) b[j] = p[j];
  return b;
}

// ---- B fragment from a [K x N] row-major f16 matrix (e.g. V) ----
static __device__ __forceinline__ v16h load_bkn_f16(const _Float16* base, int ld) {
  const int lane = threadIdx.x & 31;
  const int n = lane & 15, hf = lane >> 4;
  const _Float16* p = base + (size_t)(16 * hf) * ld + n;
  v16h b;
#pragma unroll
  for (int j = 0; j < 16; ++j) b[j] = p[(size_t)j * ld];
  return b;
}

// ---- stage a 64x32 f16 weight tile into LDS ([64][WPAD] halves) ----
constexpr int WPAD = 40;  // 80B row stride: 16B-aligned, bank-friendly
static __device__ __forceinline__ void stage_w(_Float16* dst, const _Float16* src,
                                               int ldsrc) {
  const int t   = threadIdx.x;        // 0..255, one b128 each
  const int row = t >> 2;             // 0..63
  const int col = (t & 3) * 8;        // 0,8,16,24
  const _Float16* g = src + (size_t)row * ldsrc + col;
  _Float16*       l = dst + row * WPAD + col;
#if HAVE_ASYNC_LDS
  __builtin_amdgcn_global_load_async_to_lds_b128((gas_v4i*)g, (las_v4i*)l, 0, 0);
#else
  *(v8h*)l = *(const v8h*)g;          // load + ds_store fallback
#endif
}

// =====================================================================
// Kernel 0: fp32 -> f16 conversion pre-pass (one shot, removes all
// v_cvt work from the GEMM hot loops)
// =====================================================================
__global__ __launch_bounds__(256) void cvt_f16_kernel(const float* __restrict__ src,
                                                      _Float16* __restrict__ dst,
                                                      int n8) {
  const int i = (blockIdx.x * 256 + threadIdx.x) * 8;
  if (i < n8) {
    const v8f v = *(const v8f*)(src + i);
    v8h o;
#pragma unroll
    for (int j = 0; j < 8; ++j) o[j] = (_Float16)v[j];
    *(v8h*)(dst + i) = o;
  }
}

// =====================================================================
// Kernel 1: Q/K/V projection.  out = x @ W^T (all f16 in, f16 out,
// layout [B][H][T][HS]).  grid (BT/128, D/64, 3), block 256.
// Weight tile double-buffered in LDS via async copy.
// =====================================================================
__global__ __launch_bounds__(256) void qkv_proj_kernel(
    const _Float16* __restrict__ xh,
    const _Float16* __restrict__ wh,   // [3][D][D] f16 (q,k,v)
    _Float16* __restrict__ qkvh) {
  __shared__ _Float16 Wt[2][64 * WPAD];

  const int which = blockIdx.z;
  const _Float16* W = wh + (size_t)which * DD;
  _Float16* outp = qkvh + (size_t)which * BTD;

  const int wave = threadIdx.x >> 5;
  const int lane = threadIdx.x & 31;
  const int row0 = blockIdx.x * 128 + wave * 16;
  const int col0 = blockIdx.y * 64;

  const _Float16* Wbase = W + (size_t)col0 * Dd;
  stage_w(Wt[0], Wbase, Dd);

  v8f acc[4] = {v8f{}, v8f{}, v8f{}, v8f{}};
  constexpr int NK = Dd / 32;
  for (int kk = 0; kk < NK; ++kk) {
    const int cur = kk & 1;
    wait_async0();
    __syncthreads();                            // buffer `cur` visible to all
    if (kk + 1 < NK) stage_w(Wt[cur ^ 1], Wbase + (kk + 1) * 32, Dd);
    __builtin_prefetch(xh + (size_t)row0 * Dd + (kk + 2) * 32, 0, 0);
    v16h a = load_a_f16(xh + (size_t)row0 * Dd + kk * 32, Dd);
#pragma unroll
    for (int t = 0; t < 4; ++t) {
      v16h b = load_bt_f16(&Wt[cur][16 * t * WPAD], WPAD);
      acc[t] = wmma16(a, b, acc[t]);
    }
  }

  const int hf = lane >> 4, nl = lane & 15;
#pragma unroll
  for (int t = 0; t < 4; ++t) {
    const int n = col0 + 16 * t + nl;
    const int h = n >> 6, hs = n & 63;
#pragma unroll
    for (int r = 0; r < 8; ++r) {
      const int grow = row0 + 8 * hf + r;       // 0..BT-1
      const int bb = grow >> 11, tt = grow & (Tt - 1);
      outp[(((size_t)bb * Hh + h) * Tt + tt) * HS + hs] = (_Float16)acc[t][r];
    }
  }
}

// =====================================================================
// Kernel 2: fused scores + softmax + P@V.
// grid (B*H, T/16), block 256.  Wave w owns key chunk [w*256, w*256+256).
// Writes att (fp32, normalized) exactly once; y_pre (f16) to workspace.
// =====================================================================
__global__ __launch_bounds__(256) void attn_kernel(
    const _Float16* __restrict__ qkvh,
    float* __restrict__ att,
    _Float16* __restrict__ yph) {
  __shared__ float    redbuf[8][16];            // per-wave row stats
  __shared__ float    yacc[16][64];             // cross-wave y accumulator
  __shared__ _Float16 Pstage[8][16][32];        // wave-private P re-layout tile

  const int bh   = blockIdx.x;                  // b*H + h
  const int q0   = blockIdx.y * 16;
  const int wave = threadIdx.x >> 5;
  const int lane = threadIdx.x & 31;
  const int hf   = lane >> 4, nl = lane & 15;

  const _Float16* Qh = qkvh + ((size_t)bh * Tt + q0) * HS;
  const _Float16* Kh = qkvh + BTD + (size_t)bh * Tt * HS;
  const _Float16* Vh = qkvh + 2 * BTD + (size_t)bh * Tt * HS;

  // zero the cross-wave accumulator (first __syncthreads covers it)
  for (int i = threadIdx.x; i < 16 * 64; i += 256) (&yacc[0][0])[i] = 0.f;

  // Q fragments: K-dim = HS = 64 -> two 32-wide steps
  const v16h aq0 = load_a_f16(Qh + 0, HS);
  const v16h aq1 = load_a_f16(Qh + 32, HS);

  const int kb0 = wave * 256;

  // ---- S = (Q K^T) / sqrt(HS) for this wave's 16x256 chunk ----
  v8f s[16];
#pragma unroll
  for (int nt = 0; nt < 16; ++nt) {
    const _Float16* kp = Kh + (size_t)(kb0 + nt * 16) * HS;
    __builtin_prefetch(kp + 16 * HS, 0, 0);
    v8f acc = v8f{};
    acc = wmma16(aq0, load_bt_f16(kp + 0, HS), acc);
    acc = wmma16(aq1, load_bt_f16(kp + 32, HS), acc);
#pragma unroll
    for (int r = 0; r < 8; ++r) acc[r] *= 0.125f;   // 1/sqrt(64)
    s[nt] = acc;
  }

  // ---- row max: per-lane over 16 tiles, then across 16 lanes (same half) ----
  float pm[8];
#pragma unroll
  for (int r = 0; r < 8; ++r) {
    float m = s[0][r];
#pragma unroll
    for (int nt = 1; nt < 16; ++nt) m = fmaxf(m, s[nt][r]);
#pragma unroll
    for (int d = 1; d < 16; d <<= 1) m = fmaxf(m, __shfl_xor(m, d, 32));
    pm[r] = m;
  }
  if (nl == 0) {
#pragma unroll
    for (int r = 0; r < 8; ++r) redbuf[wave][8 * hf + r] = pm[r];
  }
  __syncthreads();
  float gmax[8];
#pragma unroll
  for (int r = 0; r < 8; ++r) {
    float m = -3.0e38f;
#pragma unroll
    for (int w = 0; w < 8; ++w) m = fmaxf(m, redbuf[w][8 * hf + r]);
    gmax[r] = m;
  }
  __syncthreads();  // redbuf reused for sums

  // ---- exp + row sum ----
  float ps[8];
#pragma unroll
  for (int r = 0; r < 8; ++r) ps[r] = 0.f;
#pragma unroll
  for (int nt = 0; nt < 16; ++nt) {
#pragma unroll
    for (int r = 0; r < 8; ++r) {
      const float e = __expf(s[nt][r] - gmax[r]);
      s[nt][r] = e;
      ps[r] += e;
    }
  }
#pragma unroll
  for (int r = 0; r < 8; ++r) {
#pragma unroll
    for (int d = 1; d < 16; d <<= 1) ps[r] += __shfl_xor(ps[r], d, 32);
  }
  if (nl == 0) {
#pragma unroll
    for (int r = 0; r < 8; ++r) redbuf[wave][8 * hf + r] = ps[r];
  }
  __syncthreads();
  float inv[8];
#pragma unroll
  for (int r = 0; r < 8; ++r) {
    float sum = 0.f;
#pragma unroll
    for (int w = 0; w < 8; ++w) sum += redbuf[w][8 * hf + r];
    inv[r] = __frcp_rn(sum);
  }

  // ---- normalize; write att (fp32) exactly once ----
  float* attrow = att + ((size_t)bh * Tt + q0) * Tt;
#pragma unroll
  for (int nt = 0; nt < 16; ++nt) {
    const int key = kb0 + nt * 16 + nl;
#pragma unroll
    for (int r = 0; r < 8; ++r) {
      const float p = s[nt][r] * inv[r];
      s[nt][r] = p;
      attrow[(size_t)(8 * hf + r) * Tt + key] = p;
    }
  }

  // ---- P @ V over this wave's key chunk (8 k-steps of 32) ----
  v8f yv[4] = {v8f{}, v8f{}, v8f{}, v8f{}};
  for (int kk = 0; kk < 8; ++kk) {
    // re-layout the two 16x16 S tiles (C-layout regs) into A-layout via LDS
#pragma unroll
    for (int sub = 0; sub < 2; ++sub) {
#pragma unroll
      for (int r = 0; r < 8; ++r)
        Pstage[wave][8 * hf + r][16 * sub + nl] = (_Float16)s[2 * kk + sub][r];
    }
    const v16h ap = load_a_f16(&Pstage[wave][0][0], 32);
    const _Float16* vb = Vh + (size_t)(kb0 + kk * 32) * HS;
#pragma unroll
    for (int t = 0; t < 4; ++t) {
      v16h b = load_bkn_f16(vb + 16 * t, HS);
      yv[t] = wmma16(ap, b, yv[t]);
    }
  }

  // cross-wave reduction of the 16x64 partial outputs (ds_add_f32)
#pragma unroll
  for (int t = 0; t < 4; ++t) {
#pragma unroll
    for (int r = 0; r < 8; ++r)
      atomicAdd(&yacc[8 * hf + r][16 * t + nl], yv[t][r]);
  }
  __syncthreads();

  // write y_pre (f16) [B][H][T][HS]
  for (int i = threadIdx.x; i < 16 * 64; i += 256) {
    const int m = i >> 6, c = i & 63;
    yph[((size_t)bh * Tt + q0 + m) * HS + c] = (_Float16)yacc[m][c];
  }
}

// =====================================================================
// Kernel 3: y = y_pre @ Wo^T + bo  (fp32 out), Wo tile via async LDS
// grid (BT/128, D/64), block 256
// =====================================================================
__global__ __launch_bounds__(256) void out_proj_kernel(
    const _Float16* __restrict__ yph,
    const _Float16* __restrict__ woh,
    const float* __restrict__ bo,
    float* __restrict__ y) {
  __shared__ _Float16 Wt[2][64 * WPAD];

  const int wave = threadIdx.x >> 5;
  const int lane = threadIdx.x & 31;
  const int row0 = blockIdx.x * 128 + wave * 16;
  const int col0 = blockIdx.y * 64;
  const int bb = row0 >> 11, t0 = row0 & (Tt - 1);

  const _Float16* Wbase = woh + (size_t)col0 * Dd;
  stage_w(Wt[0], Wbase, Dd);

  v8f acc[4] = {v8f{}, v8f{}, v8f{}, v8f{}};
  constexpr int NK = Dd / 32;
  for (int kk = 0; kk < NK; ++kk) {
    const int cur = kk & 1;
    wait_async0();
    __syncthreads();
    if (kk + 1 < NK) stage_w(Wt[cur ^ 1], Wbase + (kk + 1) * 32, Dd);
    // A element (m,k): k = h*64 + hs ; 32-blocks never cross a head boundary
    const int k0 = kk * 32;
    const int h = k0 >> 6, hs0 = k0 & 63;
    const _Float16* ab = yph + (((size_t)bb * Hh + h) * Tt + t0) * HS + hs0;
    v16h a = load_a_f16(ab, HS);
#pragma unroll
    for (int t = 0; t < 4; ++t) {
      v16h b = load_bt_f16(&Wt[cur][16 * t * WPAD], WPAD);
      acc[t] = wmma16(a, b, acc[t]);
    }
  }

  const int hf = lane >> 4, nl = lane & 15;
#pragma unroll
  for (int t = 0; t < 4; ++t) {
    const int n = col0 + 16 * t + nl;
    const float bias = bo[n];
#pragma unroll
    for (int r = 0; r < 8; ++r) {
      const int grow = row0 + 8 * hf + r;
      y[(size_t)grow * Dd + n] = acc[t][r] + bias;
    }
  }
}

// =====================================================================
extern "C" void kernel_launch(void* const* d_in, const int* in_sizes, int n_in,
                              void* d_out, int out_size, void* d_ws, size_t ws_size,
                              hipStream_t stream) {
  (void)in_sizes; (void)n_in; (void)out_size; (void)ws_size;
  const float* x  = (const float*)d_in[0];
  const float* Wq = (const float*)d_in[1];
  const float* Wk = (const float*)d_in[2];
  const float* Wv = (const float*)d_in[3];
  const float* Wo = (const float*)d_in[4];
  const float* bo = (const float*)d_in[5];

  float* y   = (float*)d_out;                 // [B,T,D]
  float* att = (float*)d_out + BTD;           // [B,H,T,T]

  // workspace layout (halves): xh | Wq,Wk,Wv,Wo (f16) | Q,K,V (f16) | y_pre
  _Float16* xh   = (_Float16*)d_ws;
  _Float16* wh   = xh + BTD;                  // 4 * DD
  _Float16* qkvh = wh + 4 * DD;               // 3 * BTD
  _Float16* yph  = qkvh + 3 * BTD;            // BTD

  // one-shot fp32->f16 conversions
  cvt_f16_kernel<<<dim3((int)(BTD / (8 * 256))), 256, 0, stream>>>(x, xh, (int)BTD);
  cvt_f16_kernel<<<dim3((int)(DD / (8 * 256))), 256, 0, stream>>>(Wq, wh + 0 * DD, (int)DD);
  cvt_f16_kernel<<<dim3((int)(DD / (8 * 256))), 256, 0, stream>>>(Wk, wh + 1 * DD, (int)DD);
  cvt_f16_kernel<<<dim3((int)(DD / (8 * 256))), 256, 0, stream>>>(Wv, wh + 2 * DD, (int)DD);
  cvt_f16_kernel<<<dim3((int)(DD / (8 * 256))), 256, 0, stream>>>(Wo, wh + 3 * DD, (int)DD);

  qkv_proj_kernel<<<dim3(BT / 128, Dd / 64, 3), 256, 0, stream>>>(xh, wh, qkvh);
  attn_kernel<<<dim3(Bb * Hh, Tt / 16), 256, 0, stream>>>(qkvh, att, yph);
  out_proj_kernel<<<dim3(BT / 128, Dd / 64), 256, 0, stream>>>(yph, wh + 3 * DD, bo, y);
}
